// MVP2MNet_58059367907674
// MI455X (gfx1250) — matrix operand order repaired and checked by reading, not compiled, not run on hardware
//
#include <hip/hip_runtime.h>
#include <hip/hip_bf16.h>
#include <math.h>

// ---------------------------------------------------------------------------
// CDNA5 (gfx1250) implementation of the Pixel2Mesh-style forward pass.
// Dense math (conv-as-GEMM + all GCN matmuls) runs on v_wmma_f32_16x16x32_bf16
// (bf16 inputs, f32 accumulate). A-tiles are streamed into LDS with the CDNA5
// async copy engine (global_load_async_to_lds_b128 + s_wait_asynccnt) with
// double buffering. Sparse segment-sum uses global f32 atomics.
// ---------------------------------------------------------------------------

typedef __attribute__((ext_vector_type(16))) __bf16 v16bf;
typedef __attribute__((ext_vector_type(8)))  float  v8f;

#define GTILE_M 128
#define GTILE_N 64
#define GTILE_K 32
#define HIDc    192
#define FEATc   963

// ============================ WMMA GEMM ====================================
// C[M,N] = A[M,K] * B[K,N] (+bias, optional relu).
// A bf16 row-major (lda mult of 32, rows 16B aligned), B bf16 row-major,
// C f32. K must be a multiple of 32 (buffers are zero-padded).
// 8 waves/block: wave grid 4(M) x 2(N), 32x32 per wave via 2x2 WMMA tiles.
// Double-buffered LDS; A staged by the async engine, B staged transposed.
__global__ __launch_bounds__(256) void k_gemm_bf16_wmma(
    const __bf16* __restrict__ A, int lda,
    const __bf16* __restrict__ B, int ldb,
    float* __restrict__ C, int ldc,
    const float* __restrict__ bias,
    int M, int N, int K, int relu)
{
  __shared__ __bf16 As[2][GTILE_M][GTILE_K];   // 2 x 8 KB
  __shared__ __bf16 Bs[2][GTILE_N][GTILE_K];   // 2 x 4 KB, transposed [N][K]

  const int tid  = threadIdx.x;
  const int lane = tid & 31;
  const int wave = tid >> 5;
  const int wm   = wave >> 1;      // 0..3
  const int wn   = wave & 1;       // 0..1
  const int l16  = lane & 15;
  const int half = lane >> 4;
  const int m0   = blockIdx.x * GTILE_M;
  const int n0   = blockIdx.y * GTILE_N;

  v8f acc[2][2];
#pragma unroll
  for (int i = 0; i < 2; ++i)
#pragma unroll
    for (int j = 0; j < 2; ++j)
#pragma unroll
      for (int r = 0; r < 8; ++r) acc[i][j][r] = 0.0f;

  union Frag { uint4 u[2]; v16bf v; };

  const int arow = tid >> 1;        // 0..127
  const int aseg = tid & 1;         // 16 bf16 (32B) per thread
  const int bn   = tid & 63;        // B stage: column within tile
  const int bk0  = (tid >> 6) * 8;  // 8 K-rows per thread

  const bool a_full = (m0 + GTILE_M <= M);   // uniform across block

  // ---- tile staging ----
  auto stage_A = [&](int k0, int buf) {
    __bf16* dst = &As[buf][arow][aseg * 16];
    if (a_full) {
      // CDNA5 async copy (global memory -> LDS); the immediate offset applies
      // to both sides, so two b128s cover this thread's 32-byte slice.
      unsigned loff = (unsigned)(unsigned long long)(const void*)dst;
      unsigned long long gaddr =
          (unsigned long long)(const void*)(A + (size_t)(m0 + arow) * lda + k0 + aseg * 16);
      asm volatile("global_load_async_to_lds_b128 %0, %1, off"
                   :: "v"(loff), "v"(gaddr) : "memory");
      asm volatile("global_load_async_to_lds_b128 %0, %1, off offset:16"
                   :: "v"(loff), "v"(gaddr) : "memory");
    } else {
      uint4 va0 = make_uint4(0, 0, 0, 0), va1 = make_uint4(0, 0, 0, 0);
      if (m0 + arow < M) {
        const uint4* p = (const uint4*)(A + (size_t)(m0 + arow) * lda + k0 + aseg * 16);
        va0 = p[0]; va1 = p[1];
      }
      *(uint4*)(dst)     = va0;
      *(uint4*)(dst + 8) = va1;
    }
  };
  auto stage_B = [&](int k0, int buf) {
#pragma unroll
    for (int i = 0; i < 8; ++i) {
      __bf16 v = (__bf16)0.0f;
      if (n0 + bn < N) v = B[(size_t)(k0 + bk0 + i) * ldb + (n0 + bn)];
      Bs[buf][bn][bk0 + i] = v;
    }
  };

  const int nk = K / GTILE_K;
  stage_A(0, 0);
  stage_B(0, 0);
  asm volatile("s_wait_asynccnt 0x0" ::: "memory");
  __syncthreads();

  for (int kb = 0; kb < nk; ++kb) {
    const int cur = kb & 1;
    if (kb + 1 < nk) {           // fill the other buffer while computing
      stage_A((kb + 1) * GTILE_K, cur ^ 1);
      if (n0 + bn < N)           // gfx1250 global_prefetch_b8 on the B stream
        __builtin_prefetch(B + (size_t)((kb + 1) * GTILE_K + bk0) * ldb + n0 + bn, 0, 1);
      stage_B((kb + 1) * GTILE_K, cur ^ 1);
    }

    // ---- fragments per documented CDNA5 bf16 layouts ----
    // A 16x32: lane = M%16 | half; K pairs at byte offsets {0..12,32..44}+half*16
    Frag a[2], b[2];
#pragma unroll
    for (int i = 0; i < 2; ++i) {
      const char* pa = (const char*)&As[cur][wm * 32 + i * 16 + l16][0] + half * 16;
      a[i].u[0] = *(const uint4*)(pa);
      a[i].u[1] = *(const uint4*)(pa + 32);
    }
    // B 32x16 (transposed in LDS): lane = N%16 | half; K pairs at byte offsets {0..12,32..44}+half*16
#pragma unroll
    for (int j = 0; j < 2; ++j) {
      const char* pb = (const char*)&Bs[cur][wn * 32 + j * 16 + l16][0] + half * 16;
      b[j].u[0] = *(const uint4*)(pb);
      b[j].u[1] = *(const uint4*)(pb + 32);
    }
#pragma unroll
    for (int i = 0; i < 2; ++i)
#pragma unroll
      for (int j = 0; j < 2; ++j)
        acc[i][j] = __builtin_amdgcn_wmma_f32_16x16x32_bf16(
            false, a[i].v, false, b[j].v, (short)0, acc[i][j], false, false);

    asm volatile("s_wait_asynccnt 0x0" ::: "memory");  // my async fills done
    __syncthreads();                                   // publish to workgroup
  }

  // ---- epilogue: C/D layout: VGPR r -> row = r + 8*half, col = l16 ----
#pragma unroll
  for (int i = 0; i < 2; ++i) {
#pragma unroll
    for (int j = 0; j < 2; ++j) {
      int col = n0 + wn * 32 + j * 16 + l16;
      if (col >= N) continue;
      float bv = bias ? bias[col] : 0.0f;
#pragma unroll
      for (int r = 0; r < 8; ++r) {
        int row = m0 + wm * 32 + i * 16 + half * 8 + r;
        if (row < M) {
          float v = acc[i][j][r] + bv;
          if (relu) v = fmaxf(v, 0.0f);
          C[(size_t)row * ldc + col] = v;
        }
      }
    }
  }
}

// ========================= data-movement kernels ===========================

// f32 [M][Ksrc] -> bf16 [M][Kdst] (zero-pad extra cols)
__global__ void k_f32_to_bf16(const float* __restrict__ src, __bf16* __restrict__ dst,
                              int M, int Ksrc, int Kdst)
{
  long t = (long)blockIdx.x * blockDim.x + threadIdx.x;
  long total = (long)M * Kdst;
  if (t >= total) return;
  int c = (int)(t % Kdst);
  long r = t / Kdst;
  dst[t] = (c < Ksrc) ? (__bf16)src[r * Ksrc + c] : (__bf16)0.0f;
}

// f32 [K][N] -> bf16 [Kpad][N] (zero-pad extra rows)
__global__ void k_padrows_bf16(const float* __restrict__ src, __bf16* __restrict__ dst,
                               int K, int N, int Kpad)
{
  long t = (long)blockIdx.x * blockDim.x + threadIdx.x;
  long total = (long)Kpad * N;
  if (t >= total) return;
  int n = (int)(t % N);
  int k = (int)(t / N);
  dst[t] = (k < K) ? (__bf16)src[(size_t)k * N + n] : (__bf16)0.0f;
}

// conv weight (Cout,Cin,kh,kw) f32 -> bf16 [Kpad][Cout]; k = ci*KK + kh*Kk + kw
__global__ void k_convw(const float* __restrict__ w, __bf16* __restrict__ dst,
                        int Cout, int Cin, int Kk, int Kpad)
{
  long t = (long)blockIdx.x * blockDim.x + threadIdx.x;
  long total = (long)Kpad * Cout;
  if (t >= total) return;
  int co = (int)(t % Cout);
  int k  = (int)(t / Cout);
  int KK = Kk * Kk;
  float v = 0.0f;
  if (k < Cin * KK) {
    int ci = k / KK, r = k % KK;
    v = w[(size_t)(co * Cin + ci) * KK + r];
  }
  dst[t] = (__bf16)v;
}

// im2col: src NHWC f32 (or NCHW if nchw=1) -> bf16 [3*Ho*Wo][Kpad]
__global__ void k_im2col(const float* __restrict__ src, int nchw, int Himg, int Wimg,
                         int Cin, int Ho, int Wo, int Kk, int stride, int pad_t, int pad_l,
                         __bf16* __restrict__ dst, int Kpad)
{
  long t = (long)blockIdx.x * blockDim.x + threadIdx.x;
  long total = (long)3 * Ho * Wo * Kpad;
  if (t >= total) return;
  int c = (int)(t % Kpad);
  long row = t / Kpad;
  int wo = (int)(row % Wo);
  long tmp = row / Wo;
  int ho = (int)(tmp % Ho);
  int n  = (int)(tmp / Ho);
  int KK = Kk * Kk;
  float v = 0.0f;
  if (c < Cin * KK) {
    int ci = c / KK, r = c % KK, kh = r / Kk, kw = r % Kk;
    int h = ho * stride + kh - pad_t;
    int w = wo * stride + kw - pad_l;
    if (h >= 0 && h < Himg && w >= 0 && w < Wimg) {
      v = nchw ? src[(((size_t)n * Cin + ci) * Himg + h) * Wimg + w]
               : src[(((size_t)n * Himg + h) * Wimg + w) * Cin + ci];
    }
  }
  dst[t] = (__bf16)v;
}

// sparse scatter: z[idx[e][0]][f] += vals[e] * y[idx[e][1]][f]
__global__ void k_spmm(const int* __restrict__ idx, const float* __restrict__ vals,
                       const float* __restrict__ y, float* __restrict__ z, int E, int F)
{
  long t = (long)blockIdx.x * blockDim.x + threadIdx.x;
  long total = (long)E * F;
  if (t >= total) return;
  int f = (int)(t % F);
  int e = (int)(t / F);
  int dstr = idx[(size_t)e * 2 + 0];
  int srcr = idx[(size_t)e * 2 + 1];
  atomicAdd(&z[(size_t)dstr * F + f], vals[e] * y[(size_t)srcr * F + f]);
}

__global__ void k_bias_act(const float* __restrict__ z, const float* __restrict__ bias,
                           int M, int F, int relu, float* __restrict__ out)
{
  long t = (long)blockIdx.x * blockDim.x + threadIdx.x;
  long total = (long)M * F;
  if (t >= total) return;
  int c = (int)(t % F);
  float v = z[t] + bias[c];
  if (relu) v = fmaxf(v, 0.0f);
  out[t] = v;
}

__global__ void k_residual(float* __restrict__ h, const float* __restrict__ prev, long total)
{
  long t = (long)blockIdx.x * blockDim.x + threadIdx.x;
  if (t >= total) return;
  h[t] = 0.5f * (h[t] + prev[t]);
}

// graph unpool for 3-wide coords: dst[0:n]=src; dst[n+j]=mean(src[i0],src[i1])
__global__ void k_gpool3(const float* __restrict__ src, const int* __restrict__ pidx,
                         float* __restrict__ dst, int n, int m)
{
  long t = (long)blockIdx.x * blockDim.x + threadIdx.x;
  long total = (long)(n + m) * 3;
  if (t >= total) return;
  int c = (int)(t % 3);
  int row = (int)(t / 3);
  if (row < n) { dst[t] = src[t]; return; }
  int j = row - n;
  int i0 = pidx[(size_t)j * 2 + 0], i1 = pidx[(size_t)j * 2 + 1];
  dst[t] = 0.5f * (src[(size_t)i0 * 3 + c] + src[(size_t)i1 * 3 + c]);
}

// graph unpool of hidden (192-wide) written as bf16 into xmat[:, coloff:coloff+192]
__global__ void k_gpool_hid(const float* __restrict__ hid, const int* __restrict__ pidx,
                            __bf16* __restrict__ xm, int ldx, int coloff, int n, int m)
{
  long t = (long)blockIdx.x * blockDim.x + threadIdx.x;
  long total = (long)(n + m) * HIDc;
  if (t >= total) return;
  int c = (int)(t % HIDc);
  int row = (int)(t / HIDc);
  float v;
  if (row < n) v = hid[(size_t)row * HIDc + c];
  else {
    int j = row - n;
    int i0 = pidx[(size_t)j * 2 + 0], i1 = pidx[(size_t)j * 2 + 1];
    v = 0.5f * (hid[(size_t)i0 * HIDc + c] + hid[(size_t)i1 * HIDc + c]);
  }
  xm[(size_t)row * ldx + coloff + c] = (__bf16)v;
}

// camera precompute: per view, cm (3x3 row-major, normalized) + unnormalized Zv
__global__ void k_campre(const float* __restrict__ cams, float* __restrict__ out)
{
  int v = threadIdx.x;
  if (v >= 3) return;
  const float PI = 3.14159265358979323846f;
  const float* c = cams + v * 5;
  float th = c[0] * (PI / 180.0f), ph = c[1] * (PI / 180.0f);
  float camy = c[3] * sinf(ph), lens = c[3] * cosf(ph);
  float Zv[3] = {lens * cosf(th), camy, lens * sinf(th)};
  float Yv[3] = {camy * cosf(th + PI), lens, camy * sinf(th + PI)};
  float Xv[3] = {Yv[1] * Zv[2] - Yv[2] * Zv[1],
                 Yv[2] * Zv[0] - Yv[0] * Zv[2],
                 Yv[0] * Zv[1] - Yv[1] * Zv[0]};
  float* o = out + v * 12;
  o[9] = Zv[0]; o[10] = Zv[1]; o[11] = Zv[2];
  float rows[3][3] = {{Xv[0],Xv[1],Xv[2]}, {Yv[0],Yv[1],Yv[2]}, {Zv[0],Zv[1],Zv[2]}};
  for (int r = 0; r < 3; ++r) {
    float nn = sqrtf(rows[r][0]*rows[r][0] + rows[r][1]*rows[r][1] + rows[r][2]*rows[r][2]) + 1e-8f;
    o[r*3+0] = rows[r][0] / nn; o[r*3+1] = rows[r][1] / nn; o[r*3+2] = rows[r][2] / nn;
  }
}

// perceptual projection: one block per vertex, 960 threads = one per feature.
// writes bf16 cols [0..2]=coords, [3..962]=mean-over-views bilinear samples.
__global__ __launch_bounds__(960) void k_project(
    const float* __restrict__ coords, const float* __restrict__ cam,
    const float* __restrict__ f2, const float* __restrict__ f3,
    const float* __restrict__ f4, const float* __restrict__ f5,
    __bf16* __restrict__ xmat, int ldx)
{
  __shared__ float shh[3], sww[3];
  int vid = blockIdx.x;
  int f = threadIdx.x;
  float cx = coords[(size_t)vid * 3 + 0];
  float cy = coords[(size_t)vid * 3 + 1];
  float cz = coords[(size_t)vid * 3 + 2];
  if (f < 3) {
    const float* cd = cam + f * 12;
    float px = cx - cd[9], py = cy - cd[10], pz = cz - cd[11];
    float X = cd[0]*px + cd[1]*py + cd[2]*pz;
    float Y = cd[3]*px + cd[4]*py + cd[5]*pz;
    float Z = cd[6]*px + cd[7]*py + cd[8]*pz;
    float depth = -Z;
    depth = (depth >= 0.0f) ? fmaxf(depth, 0.1f) : fminf(depth, -0.1f);
    float hh = 248.0f * (-Y) / depth + 112.0f;
    float ww = 248.0f * X / depth + 112.0f;
    shh[f] = fminf(fmaxf(hh, 0.0f), 223.0f);
    sww[f] = fminf(fmaxf(ww, 0.0f), 223.0f);
    xmat[(size_t)vid * ldx + f] = (__bf16)coords[(size_t)vid * 3 + f];
  }
  __syncthreads();

  const float* fm; int H, C, cch;
  if (f < 64)       { fm = f2; H = 56; C = 64;  cch = f; }
  else if (f < 192) { fm = f3; H = 28; C = 128; cch = f - 64; }
  else if (f < 448) { fm = f4; H = 14; C = 256; cch = f - 192; }
  else              { fm = f5; H = 7;  C = 512; cch = f - 448; }

  float accv = 0.0f;
#pragma unroll
  for (int v = 0; v < 3; ++v) {
    float sc = (float)H / 224.0f;
    float x = fminf(fmaxf(shh[v] * sc, 0.0f), (float)H - 1.0f);
    float y = fminf(fmaxf(sww[v] * sc, 0.0f), (float)H - 1.0f);
    float x0 = floorf(x), y0 = floorf(y);
    float fx = x - x0, fy = y - y0;
    int x0i = (int)x0, y0i = (int)y0;
    int x1i = x0i + 1 < H ? x0i + 1 : H - 1;
    int y1i = y0i + 1 < H ? y0i + 1 : H - 1;
    const float* basep = fm + (size_t)v * H * H * C;   // NHWC, image v
    float v00 = basep[((size_t)x0i * H + y0i) * C + cch];
    float v01 = basep[((size_t)x0i * H + y1i) * C + cch];
    float v10 = basep[((size_t)x1i * H + y0i) * C + cch];
    float v11 = basep[((size_t)x1i * H + y1i) * C + cch];
    accv += v00 * (1.0f - fx) * (1.0f - fy) + v01 * (1.0f - fx) * fy
          + v10 * fx * (1.0f - fy)          + v11 * fx * fy;
  }
  xmat[(size_t)vid * ldx + 3 + f] = (__bf16)(accv * (1.0f / 3.0f));
}

// ============================ host orchestration ===========================

extern "C" void kernel_launch(void* const* d_in, const int* in_sizes, int n_in,
                              void* d_out, int out_size, void* d_ws, size_t ws_size,
                              hipStream_t stream)
{
  (void)in_sizes; (void)n_in; (void)out_size; (void)ws_size;
  const int N1v = 2562, N2v = 10242, N3v = 40962;
  const int E1 = 18000, E2 = 72000, E3 = 287000;
  const int M1p = N2v - N1v, M2p = N3v - N2v;

  const float* imgs   = (const float*)d_in[0];
  const float* icoord = (const float*)d_in[1];
  const float* cams   = (const float*)d_in[2];
  const int*   s1i = (const int*)d_in[3];  const float* s1v = (const float*)d_in[4];
  const int*   s2i = (const int*)d_in[5];  const float* s2v = (const float*)d_in[6];
  const int*   s3i = (const int*)d_in[7];  const float* s3v = (const float*)d_in[8];
  const int*   p1  = (const int*)d_in[9];  const int*   p2  = (const int*)d_in[10];
  // cnn_ws: d_in[11..28], cnn_bs: d_in[29..46]
  const float* g1w0 = (const float*)d_in[47]; const float* g1b0 = (const float*)d_in[48];
  const float* g1wm = (const float*)d_in[49]; const float* g1bm = (const float*)d_in[50];
  const float* g1wo = (const float*)d_in[51]; const float* g1bo = (const float*)d_in[52];
  const float* g2w0 = (const float*)d_in[53]; const float* g2b0 = (const float*)d_in[54];
  const float* g2wm = (const float*)d_in[55]; const float* g2bm = (const float*)d_in[56];
  const float* g2wo = (const float*)d_in[57]; const float* g2bo = (const float*)d_in[58];
  const float* g3w0 = (const float*)d_in[59]; const float* g3b0 = (const float*)d_in[60];
  const float* g3wm = (const float*)d_in[61]; const float* g3bm = (const float*)d_in[62];
  const float* g3wo = (const float*)d_in[63]; const float* g3bo = (const float*)d_in[64];

  // --------- workspace arena ---------
  char* base = (char*)d_ws;
  size_t off = 0;
  auto alloc = [&](size_t bytes) -> char* {
    off = (off + 255) & ~(size_t)255;
    char* p = base + off;
    off += bytes;
    return p;
  };
  auto rup32 = [](int x) { return (x + 31) & ~31; };
  auto ew = [](size_t t) { return dim3((unsigned)((t + 255) / 256)); };

  float*  camdata = (float*)alloc(3 * 12 * 4);
  __bf16* wconv   = (__bf16*)alloc((size_t)6400 * 512 * 2);
  __bf16* colbuf  = (__bf16*)alloc((size_t)150528 * 160 * 2);
  float*  actA    = (float*)alloc((size_t)150528 * 16 * 4);
  float*  actB    = (float*)alloc((size_t)150528 * 16 * 4);
  float*  f2 = (float*)alloc((size_t)3 * 56 * 56 * 64 * 4);
  float*  f3 = (float*)alloc((size_t)3 * 28 * 28 * 128 * 4);
  float*  f4 = (float*)alloc((size_t)3 * 14 * 14 * 256 * 4);
  float*  f5 = (float*)alloc((size_t)3 * 7 * 7 * 512 * 4);
  __bf16* xmat = (__bf16*)alloc((size_t)N3v * 1184 * 2);
  float*  y0   = (float*)alloc((size_t)N3v * HIDc * 4);
  float*  y1   = (float*)alloc((size_t)N3v * HIDc * 4);
  float*  zb   = (float*)alloc((size_t)N3v * HIDc * 4);
  float*  hb0  = (float*)alloc((size_t)N3v * HIDc * 4);
  float*  hb1  = (float*)alloc((size_t)N3v * HIDc * 4);
  float*  hb2  = (float*)alloc((size_t)N3v * HIDc * 4);
  __bf16* hbf  = (__bf16*)alloc((size_t)N3v * HIDc * 2);
  __bf16* wb0  = (__bf16*)alloc((size_t)1184 * HIDc * 2);
  __bf16* wb1  = (__bf16*)alloc((size_t)1184 * HIDc * 2);
  __bf16* wob0 = (__bf16*)alloc((size_t)HIDc * 3 * 2);
  __bf16* wob1 = (__bf16*)alloc((size_t)HIDc * 3 * 2);
  float*  out1  = (float*)alloc((size_t)N1v * 3 * 4);
  float*  out1p = (float*)alloc((size_t)N2v * 3 * 4);
  float*  out2  = (float*)alloc((size_t)N2v * 3 * 4);
  float*  out2p = (float*)alloc((size_t)N3v * 3 * 4);

  auto launch_gemm = [&](const __bf16* A, int lda, const __bf16* B, int ldb,
                         float* C, int ldc, const float* bias,
                         int M, int N, int K, int relu) {
    dim3 g((M + GTILE_M - 1) / GTILE_M, (N + GTILE_N - 1) / GTILE_N);
    k_gemm_bf16_wmma<<<g, 256, 0, stream>>>(A, lda, B, ldb, C, ldc, bias, M, N, K, relu);
  };

  // ============================== CNN ==============================
  struct CL { int ci, co, k, s, p; };
  const CL cls[18] = {
      {3,16,3,1,1},{16,16,3,1,1},{16,32,3,2,0},{32,32,3,1,1},{32,32,3,1,1},
      {32,64,3,2,0},{64,64,3,1,1},{64,64,3,1,1},{64,128,3,2,0},{128,128,3,1,1},
      {128,128,3,1,1},{128,256,5,2,1},{256,256,3,1,1},{256,256,3,1,1},
      {256,512,5,2,1},{512,512,3,1,1},{512,512,3,1,1},{512,512,3,1,1}};
  const float* cur = imgs;
  int curH = 224, nchw = 1, flip = 0;
  for (int li = 0; li < 18; ++li) {
    CL L = cls[li];
    int Ho = (L.s == 1) ? curH : curH / 2;
    int Kl = L.ci * L.k * L.k;
    int Kp = rup32(Kl);
    int Mg = 3 * Ho * Ho;
    const float* w  = (const float*)d_in[11 + li];
    const float* bs = (const float*)d_in[29 + li];
    k_convw<<<ew((size_t)Kp * L.co), 256, 0, stream>>>(w, wconv, L.co, L.ci, L.k, Kp);
    k_im2col<<<ew((size_t)Mg * Kp), 256, 0, stream>>>(cur, nchw, curH, curH, L.ci,
                                                      Ho, Ho, L.k, L.s, L.p, L.p, colbuf, Kp);
    float* outp;
    if (li == 7) outp = f2; else if (li == 10) outp = f3;
    else if (li == 13) outp = f4; else if (li == 17) outp = f5;
    else { outp = flip ? actB : actA; flip ^= 1; }
    launch_gemm(colbuf, Kp, wconv, L.co, outp, L.co, bs, Mg, L.co, Kp, 1);
    cur = outp; curH = Ho; nchw = 0;
  }

  k_campre<<<1, 32, 0, stream>>>(cams, camdata);

  // ============================== GCN blocks =======================
  float* bufs[3] = {hb0, hb1, hb2};
  auto run_gcn = [&](const __bf16* x, int Kx, int Kxl, int nv,
                     const float* w0, const float* b0,
                     const float* wmid, const float* bmid, int Lmid,
                     const float* wout, const float* bout,
                     const int* sidx, const float* svals, int E,
                     float* outv, float** hid_out) {
    // layer 0 (x -> 192)
    k_padrows_bf16<<<ew((size_t)Kx * HIDc), 256, 0, stream>>>(w0, wb0, Kxl, HIDc, Kx);
    k_padrows_bf16<<<ew((size_t)Kx * HIDc), 256, 0, stream>>>(w0 + (size_t)Kxl * HIDc, wb1, Kxl, HIDc, Kx);
    launch_gemm(x, Kx, wb0, HIDc, y0, HIDc, nullptr, nv, HIDc, Kx, 0);
    launch_gemm(x, Kx, wb1, HIDc, y1, HIDc, nullptr, nv, HIDc, Kx, 0);
    hipMemsetAsync(zb, 0, (size_t)nv * HIDc * 4, stream);
    k_spmm<<<ew((size_t)E * HIDc), 256, 0, stream>>>(sidx, svals, y0, zb, E, HIDc);
    k_spmm<<<ew((size_t)E * HIDc), 256, 0, stream>>>(sidx + (size_t)2 * E, svals + E, y1, zb, E, HIDc);
    k_bias_act<<<ew((size_t)nv * HIDc), 256, 0, stream>>>(zb, b0, nv, HIDc, 1, bufs[0]);
    k_f32_to_bf16<<<ew((size_t)nv * HIDc), 256, 0, stream>>>(bufs[0], hbf, nv, HIDc, HIDc);
    // mid layers
    for (int i = 1; i <= Lmid; ++i) {
      float* hc = bufs[i % 3];
      const float* wl = wmid + (size_t)(i - 1) * 2 * HIDc * HIDc;
      k_padrows_bf16<<<ew((size_t)HIDc * HIDc), 256, 0, stream>>>(wl, wb0, HIDc, HIDc, HIDc);
      k_padrows_bf16<<<ew((size_t)HIDc * HIDc), 256, 0, stream>>>(wl + (size_t)HIDc * HIDc, wb1, HIDc, HIDc, HIDc);
      launch_gemm(hbf, HIDc, wb0, HIDc, y0, HIDc, nullptr, nv, HIDc, HIDc, 0);
      launch_gemm(hbf, HIDc, wb1, HIDc, y1, HIDc, nullptr, nv, HIDc, HIDc, 0);
      hipMemsetAsync(zb, 0, (size_t)nv * HIDc * 4, stream);
      k_spmm<<<ew((size_t)E * HIDc), 256, 0, stream>>>(sidx, svals, y0, zb, E, HIDc);
      k_spmm<<<ew((size_t)E * HIDc), 256, 0, stream>>>(sidx + (size_t)2 * E, svals + E, y1, zb, E, HIDc);
      k_bias_act<<<ew((size_t)nv * HIDc), 256, 0, stream>>>(zb, bmid + (size_t)(i - 1) * HIDc, nv, HIDc, 1, hc);
      if ((i % 2) == 0)
        k_residual<<<ew((size_t)nv * HIDc), 256, 0, stream>>>(hc, bufs[(i - 2) % 3], (long)nv * HIDc);
      k_f32_to_bf16<<<ew((size_t)nv * HIDc), 256, 0, stream>>>(hc, hbf, nv, HIDc, HIDc);
    }
    // output layer (192 -> 3, no relu)
    k_padrows_bf16<<<ew((size_t)HIDc * 3), 256, 0, stream>>>(wout, wob0, HIDc, 3, HIDc);
    k_padrows_bf16<<<ew((size_t)HIDc * 3), 256, 0, stream>>>(wout + (size_t)HIDc * 3, wob1, HIDc, 3, HIDc);
    launch_gemm(hbf, HIDc, wob0, 3, y0, 3, nullptr, nv, 3, HIDc, 0);
    launch_gemm(hbf, HIDc, wob1, 3, y1, 3, nullptr, nv, 3, HIDc, 0);
    hipMemsetAsync(zb, 0, (size_t)nv * 3 * 4, stream);
    k_spmm<<<ew((size_t)E * 3), 256, 0, stream>>>(sidx, svals, y0, zb, E, 3);
    k_spmm<<<ew((size_t)E * 3), 256, 0, stream>>>(sidx + (size_t)2 * E, svals + E, y1, zb, E, 3);
    k_bias_act<<<ew((size_t)nv * 3), 256, 0, stream>>>(zb, bout, nv, 3, 0, outv);
    *hid_out = bufs[Lmid % 3];
  };

  // ---- block 1 ----
  int Kx1 = rup32(FEATc);            // 992
  hipMemsetAsync(xmat, 0, (size_t)N1v * Kx1 * 2, stream);
  k_project<<<dim3(N1v), 960, 0, stream>>>(icoord, camdata, f2, f3, f4, f5, xmat, Kx1);
  float* hid1 = nullptr;
  run_gcn(xmat, Kx1, FEATc, N1v, g1w0, g1b0, g1wm, g1bm, 12, g1wo, g1bo, s1i, s1v, E1, out1, &hid1);

  // ---- pool 1 -> block 2 ----
  k_gpool3<<<ew((size_t)N2v * 3), 256, 0, stream>>>(out1, p1, out1p, N1v, M1p);
  int Kx2 = rup32(FEATc + HIDc);     // 1184
  hipMemsetAsync(xmat, 0, (size_t)N2v * Kx2 * 2, stream);
  k_project<<<dim3(N2v), 960, 0, stream>>>(out1p, camdata, f2, f3, f4, f5, xmat, Kx2);
  k_gpool_hid<<<ew((size_t)N2v * HIDc), 256, 0, stream>>>(hid1, p1, xmat, Kx2, FEATc, N1v, M1p);
  float* hid2 = nullptr;
  run_gcn(xmat, Kx2, FEATc + HIDc, N2v, g2w0, g2b0, g2wm, g2bm, 12, g2wo, g2bo, s2i, s2v, E2, out2, &hid2);

  // ---- pool 2 -> block 3 ----
  k_gpool3<<<ew((size_t)N3v * 3), 256, 0, stream>>>(out2, p2, out2p, N2v, M2p);
  hipMemsetAsync(xmat, 0, (size_t)N3v * Kx2 * 2, stream);
  k_project<<<dim3(N3v), 960, 0, stream>>>(out2p, camdata, f2, f3, f4, f5, xmat, Kx2);
  k_gpool_hid<<<ew((size_t)N3v * HIDc), 256, 0, stream>>>(hid2, p2, xmat, Kx2, FEATc, N2v, M2p);
  float* hid3 = nullptr;
  run_gcn(xmat, Kx2, FEATc + HIDc, N3v, g3w0, g3b0, g3wm, g3bm, 13, g3wo, g3bo, s3i, s3v, E3,
          (float*)d_out, &hid3);
}
